// LocalSage_Plus_63771674411482
// MI455X (gfx1250) — matrix operand order repaired and checked by reading, not compiled, not run on hardware
//
#include <hip/hip_runtime.h>

// ---------------- types ----------------
typedef __attribute__((ext_vector_type(16))) __bf16 v16bf;
typedef __attribute__((ext_vector_type(8)))  float  v8f;

__device__ __forceinline__ unsigned short f2bf(float f) {
  unsigned int u = __builtin_bit_cast(unsigned int, f);
  u += 0x7FFFu + ((u >> 16) & 1u);          // round-to-nearest-even
  return (unsigned short)(u >> 16);
}

// ---------------- fp32 -> bf16 (row-major, K padded with zeros) ----------------
__global__ void cvt_pad_bf16(const float* __restrict__ in, unsigned short* __restrict__ out,
                             int M, int K, int Kp) {
  for (size_t i = (size_t)blockIdx.x * blockDim.x + threadIdx.x; i < (size_t)M * Kp;
       i += (size_t)gridDim.x * blockDim.x) {
    const size_t r = i / Kp;
    const int k = (int)(i % Kp);
    out[i] = (k < K) ? f2bf(in[r * K + k]) : (unsigned short)0;
  }
}

// ---------------- fp32 W[K][N] -> bf16 Wt[N][Kp] transposed, zero padded ----------------
__global__ void cvt_t_bf16(const float* __restrict__ in, unsigned short* __restrict__ out,
                           int K, int N, int Kp) {
  for (size_t i = (size_t)blockIdx.x * blockDim.x + threadIdx.x; i < (size_t)N * Kp;
       i += (size_t)gridDim.x * blockDim.x) {
    const size_t n = i / Kp;
    const int k = (int)(i % Kp);
    out[i] = (k < K) ? f2bf(in[(size_t)k * N + n]) : (unsigned short)0;
  }
}

// ---------------- bf16 WMMA GEMM: C = act(A[MxK] @ Bt[NxK]^T + bias) ----------------
// K must be a multiple of 32 (zero padded). A ld = K, Bt ld = K.
#define BM 64
#define BN 64
#define BK 32

template <int ACT, bool WF32, bool WB16>
__global__ __launch_bounds__(128) void gemm_bf16_wmma(
    const unsigned short* __restrict__ A, const unsigned short* __restrict__ Bt,
    const float* __restrict__ bias, float* __restrict__ Cf,
    unsigned short* __restrict__ Cb, int M, int N, int K, int packK, int packLd) {
  __shared__ __align__(16) unsigned short As[2][BM][BK + 8];
  __shared__ __align__(16) unsigned short Bs[2][BN][BK + 8];

  const int tid  = threadIdx.x;
  const int lane = tid & 31;
  const int wave = tid >> 5;
  const int hi   = lane >> 4;     // half-wave select
  const int l15  = lane & 15;
  const int bm = blockIdx.y * BM;
  const int bn = blockIdx.x * BN;
  const int wm = (wave >> 1) * 32;   // wave's 32x32 patch
  const int wn = (wave & 1) * 32;

  // cooperative loaders: thread t owns (row = t/2, 16-elem chunk = (t&1)*16) of each 64x32 tile
  const int lrow = tid >> 1;
  const int lcol = (tid & 1) * 16;
  int arow = bm + lrow; if (arow >= M) arow = M - 1;   // branch-free edge clamp
  int brow = bn + lrow; if (brow >= N) brow = N - 1;
  const unsigned short* Aptr = A  + (size_t)arow * K + lcol;
  const unsigned short* Bptr = Bt + (size_t)brow * K + lcol;

  const int nk = K / BK;
  v8f acc[2][2] = {};
  uint4 a0, a1, b0, b1;

  // prologue: tile 0 -> regs -> LDS[0]
  a0 = *(const uint4*)(Aptr);     a1 = *(const uint4*)(Aptr + 8);
  b0 = *(const uint4*)(Bptr);     b1 = *(const uint4*)(Bptr + 8);
  *(uint4*)&As[0][lrow][lcol] = a0;  *(uint4*)&As[0][lrow][lcol + 8] = a1;
  *(uint4*)&Bs[0][lrow][lcol] = b0;  *(uint4*)&Bs[0][lrow][lcol + 8] = b1;

  for (int kt = 0; kt < nk; ++kt) {
    const int cb = kt & 1;
    __syncthreads();                       // LDS[cb] ready for all waves
    if (kt + 1 < nk) {                     // fetch next tile while WMMAs run
      const int off = (kt + 1) * BK;
      a0 = *(const uint4*)(Aptr + off);  a1 = *(const uint4*)(Aptr + off + 8);
      b0 = *(const uint4*)(Bptr + off);  b1 = *(const uint4*)(Bptr + off + 8);
      if (kt + 2 < nk) __builtin_prefetch(Aptr + (kt + 2) * BK, 0, 1);
    }

    union Frag { uint4 q[2]; v16bf v; };
    v16bf afr[2], bfr[2];
#pragma unroll
    for (int i = 0; i < 2; ++i) {
      // A 16x32 bf16 frag: lanes 0-15 K 0..7 & 16..23, lanes 16-31 K 8..15 & 24..31
      const int m  = wm + 16 * i + l15;
      const int kh = hi * 8;
      Frag f;
      f.q[0] = *(const uint4*)&As[cb][m][kh];
      f.q[1] = *(const uint4*)&As[cb][m][kh + 16];
      afr[i] = f.v;
    }
#pragma unroll
    for (int j = 0; j < 2; ++j) {
      // B 32x16 bf16 frag: lanes 0-15 K 0..7 & 16..23, lanes 16-31 K 8..15 & 24..31
      const int n  = wn + 16 * j + l15;
      const int kb = hi * 8;
      Frag f;
      f.q[0] = *(const uint4*)&Bs[cb][n][kb];
      f.q[1] = *(const uint4*)&Bs[cb][n][kb + 16];
      bfr[j] = f.v;
    }
#pragma unroll
    for (int i = 0; i < 2; ++i)
#pragma unroll
      for (int j = 0; j < 2; ++j)
        acc[i][j] = __builtin_amdgcn_wmma_f32_16x16x32_bf16(
            false, afr[i], false, bfr[j], (short)0, acc[i][j], false, false);

    if (kt + 1 < nk) {                     // stage next tile in the other buffer
      const int nb = cb ^ 1;
      *(uint4*)&As[nb][lrow][lcol] = a0;  *(uint4*)&As[nb][lrow][lcol + 8] = a1;
      *(uint4*)&Bs[nb][lrow][lcol] = b0;  *(uint4*)&Bs[nb][lrow][lcol + 8] = b1;
    }
  }

  // epilogue: C/D layout -> row = 8*hi + vgpr, col = lane&15
#pragma unroll
  for (int j = 0; j < 2; ++j) {
    const int col = bn + wn + 16 * j + l15;
    if (col >= N) continue;
    const float bv = bias ? bias[col] : 0.0f;
#pragma unroll
    for (int i = 0; i < 2; ++i) {
#pragma unroll
      for (int v = 0; v < 8; ++v) {
        const int row = bm + wm + 16 * i + hi * 8 + v;
        if (row >= M) continue;
        float x = acc[i][j][v] + bv;
        if (ACT == 1) x = fmaxf(x, 0.0f);
        if (ACT == 2) x = tanhf(x);
        if (WF32) Cf[(size_t)row * N + col] = x;
        if (WB16) {
          size_t idx;
          if (packK > 0) {  // repack [M][N] -> [M*(N/packK)][packLd] (zero-padded reshape)
            const int q = col / packK;
            const int rc = col - q * packK;
            idx = ((size_t)row * (N / packK) + q) * packLd + rc;
          } else {
            idx = (size_t)row * N + col;
          }
          Cb[idx] = f2bf(x);
        }
      }
    }
  }
}

// ---------------- SpMM: y[rows[e]] += vals[e] * x[cols[e]] (wave per nnz) ----------------
__global__ void spmm_coo(const int* __restrict__ rows, const int* __restrict__ cols,
                         const float* __restrict__ vals, const float* __restrict__ x,
                         float* __restrict__ y, int nnz, int d) {
  const int e = blockIdx.x * (blockDim.x >> 5) + (threadIdx.x >> 5);
  if (e >= nnz) return;
  const int lane = threadIdx.x & 31;
  const int r = rows[e], c = cols[e];
  const float v = vals[e];
  const float* xs = x + (size_t)c * d;
  float* ys = y + (size_t)r * d;
  for (int f = lane; f < d; f += 32) atomicAdd(&ys[f], v * xs[f]);
}

// classifier graph: original edges, weight 1 both directions
__global__ void spmm_orig(const int* __restrict__ edges, const float* __restrict__ x,
                          float* __restrict__ y, int E, int d) {
  const int e = blockIdx.x * (blockDim.x >> 5) + (threadIdx.x >> 5);
  if (e >= E) return;
  const int lane = threadIdx.x & 31;
  const int lo = edges[2 * e], hic = edges[2 * e + 1];
  for (int f = lane; f < d; f += 32) {
    atomicAdd(&y[(size_t)lo * d + f], x[(size_t)hic * d + f]);
    atomicAdd(&y[(size_t)hic * d + f], x[(size_t)lo * d + f]);
  }
}

// classifier graph: generated edges (row i/P) <-> (col N+i), weight newval[i]
__global__ void spmm_new(const float* __restrict__ newval, const float* __restrict__ x,
                         float* __restrict__ y, int NP, int P, int Nn, int d) {
  const int i = blockIdx.x * (blockDim.x >> 5) + (threadIdx.x >> 5);
  if (i >= NP) return;
  const float v = newval[i];
  if (v == 0.0f) return;
  const int lane = threadIdx.x & 31;
  const int row = i / P;
  const int col = Nn + i;
  for (int f = lane; f < d; f += 32) {
    atomicAdd(&y[(size_t)row * d + f], x[(size_t)col * d + f]);
    atomicAdd(&y[(size_t)col * d + f], x[(size_t)row * d + f]);
  }
}

// ---------------- elementwise epilogues ----------------
__global__ void bias_relu_bf16(const float* __restrict__ pre, const float* __restrict__ bias,
                               unsigned short* __restrict__ outb, size_t total, int d) {
  for (size_t i = (size_t)blockIdx.x * blockDim.x + threadIdx.x; i < total;
       i += (size_t)gridDim.x * blockDim.x) {
    const int col = (int)(i % d);
    outb[i] = f2bf(fmaxf(pre[i] + bias[col], 0.0f));
  }
}

__global__ void z_finalize(const float* __restrict__ pre, const float* __restrict__ bias,
                           const float* __restrict__ noise, float* __restrict__ z,
                           unsigned short* __restrict__ znb, size_t total, int d) {
  for (size_t i = (size_t)blockIdx.x * blockDim.x + threadIdx.x; i < total;
       i += (size_t)gridDim.x * blockDim.x) {
    const int col = (int)(i % d);
    const float v = fmaxf(pre[i] + bias[col], 0.0f);
    z[i] = v;
    znb[i] = f2bf(v + noise[i]);
  }
}

__global__ void degree_k(const float* __restrict__ z, const float* __restrict__ Wreg,
                         const float* __restrict__ breg, float* __restrict__ out, int n, int L) {
  const int i = blockIdx.x * blockDim.x + threadIdx.x;
  if (i >= n) return;
  float s = breg[0];
  for (int k = 0; k < L; ++k) s += z[(size_t)i * L + k] * Wreg[k];
  out[i] = fmaxf(s, 0.0f);
}

__global__ void fill_f32(float* __restrict__ p, float v, size_t n) {
  for (size_t i = (size_t)blockIdx.x * blockDim.x + threadIdx.x; i < n;
       i += (size_t)gridDim.x * blockDim.x)
    p[i] = v;
}

__global__ void newval_k(const float* __restrict__ degree, float* __restrict__ newval,
                         float* __restrict__ rowsum, int Nn, int P) {
  const int i = blockIdx.x * blockDim.x + threadIdx.x;
  if (i >= Nn * P) return;
  const int node = i / P, j = i % P;
  int dg = (int)degree[node];
  if (dg < 0) dg = 0;
  if (dg > P) dg = P;
  const float v = (j < dg) ? 1.0f : 0.0f;
  newval[i] = v;
  if (v != 0.0f) atomicAdd(&rowsum[node], v);
  rowsum[Nn + i] += v;  // generated node: single writer
}

__global__ void edge_rowsum(const int* __restrict__ edges, float* __restrict__ rowsum, int E) {
  const int e = blockIdx.x * blockDim.x + threadIdx.x;
  if (e >= E) return;
  atomicAdd(&rowsum[edges[2 * e]], 1.0f);
  atomicAdd(&rowsum[edges[2 * e + 1]], 1.0f);
}

template <bool B16>
__global__ void cls_finalize(const float* __restrict__ pre, const float* __restrict__ rowsum,
                             const float* __restrict__ bias, unsigned short* __restrict__ ob,
                             float* __restrict__ of, size_t total, int d) {
  for (size_t i = (size_t)blockIdx.x * blockDim.x + threadIdx.x; i < total;
       i += (size_t)gridDim.x * blockDim.x) {
    const size_t row = i / d;
    const int col = (int)(i % d);
    const float x = fmaxf(pre[i] / rowsum[row] + bias[col], 0.0f);
    if (B16) ob[i] = f2bf(x);
    else     of[i] = x;
  }
}

// ---------------- host ----------------
static inline int cdiv(long long a, long long b) { return (int)((a + b - 1) / b); }

extern "C" void kernel_launch(void* const* d_in, const int* in_sizes, int n_in,
                              void* d_out, int out_size, void* d_ws, size_t ws_size,
                              hipStream_t stream) {
  (void)n_in; (void)out_size; (void)ws_size;
  const float* feat   = (const float*)d_in[0];
  const int*   edges  = (const int*)d_in[1];
  const int*   adj_r  = (const int*)d_in[2];
  const int*   adj_c  = (const int*)d_in[3];
  const float* adj_v  = (const float*)d_in[4];
  const float* noise  = (const float*)d_in[5];
  const float* W_gc1 = (const float*)d_in[6];  const float* b_gc1 = (const float*)d_in[7];
  const float* W_gc2 = (const float*)d_in[8];  const float* b_gc2 = (const float*)d_in[9];
  const float* W_reg = (const float*)d_in[10]; const float* b_reg = (const float*)d_in[11];
  const float* W_fc1 = (const float*)d_in[12]; const float* b_fc1 = (const float*)d_in[13];
  const float* W_fc2 = (const float*)d_in[14]; const float* b_fc2 = (const float*)d_in[15];
  const float* W_flat= (const float*)d_in[16]; const float* b_flat= (const float*)d_in[17];
  const float* W_gc3 = (const float*)d_in[18]; const float* b_gc3 = (const float*)d_in[19];
  const float* W_gc4 = (const float*)d_in[20]; const float* b_gc4 = (const float*)d_in[21];

  const int Nn = 20000, F = 500, Fp = 512, H = 256, L = 64, P = 5, C = 16;
  const int PF = P * F;            // 2500
  const int NP = Nn * P;           // 100000
  const int NL = Nn + NP;          // 120000
  const int E   = in_sizes[1] / 2; // 320000
  const int NNZ = in_sizes[2];     // 660000
  const int D1 = 256, D2 = 2048;

  float* out_degree = (float*)d_out;
  float* out_gen    = out_degree + Nn;
  float* out_nc     = out_gen + (size_t)Nn * PF;

  char* ws = (char*)d_ws;
  size_t cur = 0;
  auto alloc = [&](size_t bytes) -> void* {
    cur = (cur + 255) & ~(size_t)255;
    void* p = ws + cur; cur += bytes; return p;
  };
  typedef unsigned short us;
  us* feat_b  = (us*)alloc((size_t)Nn * Fp * 2);     // [20000][512] zero padded
  us* wgc1_t  = (us*)alloc((size_t)H * Fp * 2);      // [256][512]
  us* wgc2_t  = (us*)alloc((size_t)L * H * 2);       // [64][256]
  us* wfc1_t  = (us*)alloc((size_t)D1 * L * 2);      // [256][64]
  us* wfc2_t  = (us*)alloc((size_t)D2 * D1 * 2);     // [2048][256]
  us* wflat_t = (us*)alloc((size_t)PF * D2 * 2);     // [2500][2048]
  us* wgc3_t  = (us*)alloc((size_t)H * Fp * 2);      // [256][512]
  us* wgc4_t  = (us*)alloc((size_t)C * H * 2);       // [16][256]
  us* h_b     = (us*)alloc((size_t)Nn * H * 2);
  us* zn_b    = (us*)alloc((size_t)Nn * L * 2);
  us* g1_b    = (us*)alloc((size_t)Nn * D1 * 2);
  us* g2_b    = (us*)alloc((size_t)Nn * D2 * 2);
  us* gen_b   = (us*)alloc((size_t)NP * Fp * 2);     // [100000][512] zero padded (packed G6 out)
  us* h2_b    = (us*)alloc((size_t)NL * H * 2);
  float* xw1   = (float*)alloc((size_t)Nn * H * 4);
  float* hpre  = (float*)alloc((size_t)Nn * H * 4);
  float* xw2   = (float*)alloc((size_t)Nn * L * 4);
  float* zpre  = (float*)alloc((size_t)Nn * L * 4);
  float* zf    = (float*)alloc((size_t)Nn * L * 4);
  float* xw3   = (float*)alloc((size_t)NL * H * 4);
  float* h2pre = (float*)alloc((size_t)NL * H * 4);
  float* xw4   = (float*)alloc((size_t)NL * C * 4);
  float* ncpre = (float*)alloc((size_t)NL * C * 4);
  float* rowsum= (float*)alloc((size_t)NL * 4);
  float* newval= (float*)alloc((size_t)NP * 4);

  // ---- bf16 casts: activations padded row-major, weights transposed [N][Kp] ----
  cvt_pad_bf16<<<cdiv((size_t)Nn * Fp, 256), 256, 0, stream>>>(feat, feat_b, Nn, F, Fp);
  cvt_t_bf16<<<cdiv((size_t)H * Fp, 256), 256, 0, stream>>>(W_gc1, wgc1_t, F, H, Fp);
  cvt_t_bf16<<<cdiv((size_t)L * H, 256), 256, 0, stream>>>(W_gc2, wgc2_t, H, L, H);
  cvt_t_bf16<<<cdiv((size_t)D1 * L, 256), 256, 0, stream>>>(W_fc1, wfc1_t, L, D1, L);
  cvt_t_bf16<<<cdiv((size_t)D2 * D1, 256), 256, 0, stream>>>(W_fc2, wfc2_t, D1, D2, D1);
  cvt_t_bf16<<<cdiv((size_t)PF * D2, 256), 256, 0, stream>>>(W_flat, wflat_t, D2, PF, D2);
  cvt_t_bf16<<<cdiv((size_t)H * Fp, 256), 256, 0, stream>>>(W_gc3, wgc3_t, F, H, Fp);
  cvt_t_bf16<<<cdiv((size_t)C * H, 256), 256, 0, stream>>>(W_gc4, wgc4_t, H, C, H);

  // ---- encoder layer 1: XW1 = feat @ W_gc1 ; h = relu(spmm(A, XW1) + b) ----
  gemm_bf16_wmma<0, true, false><<<dim3(cdiv(H, BN), cdiv(Nn, BM)), 128, 0, stream>>>(
      feat_b, wgc1_t, nullptr, xw1, nullptr, Nn, H, Fp, 0, 0);
  hipMemsetAsync(hpre, 0, (size_t)Nn * H * 4, stream);
  spmm_coo<<<cdiv(NNZ, 8), 256, 0, stream>>>(adj_r, adj_c, adj_v, xw1, hpre, NNZ, H);
  bias_relu_bf16<<<cdiv((size_t)Nn * H, 256), 256, 0, stream>>>(hpre, b_gc1, h_b, (size_t)Nn * H, H);

  // ---- encoder layer 2: z = relu(spmm(A, h @ W_gc2) + b) ----
  gemm_bf16_wmma<0, true, false><<<dim3(cdiv(L, BN), cdiv(Nn, BM)), 128, 0, stream>>>(
      h_b, wgc2_t, nullptr, xw2, nullptr, Nn, L, H, 0, 0);
  hipMemsetAsync(zpre, 0, (size_t)Nn * L * 4, stream);
  spmm_coo<<<cdiv(NNZ, 8), 256, 0, stream>>>(adj_r, adj_c, adj_v, xw2, zpre, NNZ, L);
  z_finalize<<<cdiv((size_t)Nn * L, 256), 256, 0, stream>>>(zpre, b_gc2, noise, zf, zn_b,
                                                            (size_t)Nn * L, L);

  // ---- degree regressor -> d_out[0:N] ----
  degree_k<<<cdiv(Nn, 256), 256, 0, stream>>>(zf, W_reg, b_reg, out_degree, Nn, L);

  // ---- mend-graph bookkeeping: rowsum = 1 (loops) + orig deg + predicted edges ----
  fill_f32<<<cdiv(NL, 256), 256, 0, stream>>>(rowsum, 1.0f, (size_t)NL);
  newval_k<<<cdiv(NP, 256), 256, 0, stream>>>(out_degree, newval, rowsum, Nn, P);
  edge_rowsum<<<cdiv(E, 256), 256, 0, stream>>>(edges, rowsum, E);

  // ---- generator MLP ----
  gemm_bf16_wmma<1, false, true><<<dim3(cdiv(D1, BN), cdiv(Nn, BM)), 128, 0, stream>>>(
      zn_b, wfc1_t, b_fc1, nullptr, g1_b, Nn, D1, L, 0, 0);
  gemm_bf16_wmma<1, false, true><<<dim3(cdiv(D2, BN), cdiv(Nn, BM)), 128, 0, stream>>>(
      g1_b, wfc2_t, b_fc2, nullptr, g2_b, Nn, D2, D1, 0, 0);
  hipMemsetAsync(gen_b, 0, (size_t)NP * Fp * 2, stream);  // zero pad cols 500..511
  gemm_bf16_wmma<2, true, true><<<dim3(cdiv(PF, BN), cdiv(Nn, BM)), 128, 0, stream>>>(
      g2_b, wflat_t, b_flat, out_gen, gen_b, Nn, PF, D2, /*packK=*/F, /*packLd=*/Fp);

  // ---- classifier layer 1: XW3 = fill_feats @ W_gc3 (two row-blocks) ----
  gemm_bf16_wmma<0, true, false><<<dim3(cdiv(H, BN), cdiv(Nn, BM)), 128, 0, stream>>>(
      feat_b, wgc3_t, nullptr, xw3, nullptr, Nn, H, Fp, 0, 0);
  gemm_bf16_wmma<0, true, false><<<dim3(cdiv(H, BN), cdiv(NP, BM)), 128, 0, stream>>>(
      gen_b, wgc3_t, nullptr, xw3 + (size_t)Nn * H, nullptr, NP, H, Fp, 0, 0);
  // spmm over mended graph: loop term = copy, then edge scatters, then D^-1 + bias + relu
  hipMemcpyAsync(h2pre, xw3, (size_t)NL * H * 4, hipMemcpyDeviceToDevice, stream);
  spmm_orig<<<cdiv(E, 8), 256, 0, stream>>>(edges, xw3, h2pre, E, H);
  spmm_new<<<cdiv(NP, 8), 256, 0, stream>>>(newval, xw3, h2pre, NP, P, Nn, H);
  cls_finalize<true><<<cdiv((size_t)NL * H, 256), 256, 0, stream>>>(
      h2pre, rowsum, b_gc3, h2_b, nullptr, (size_t)NL * H, H);

  // ---- classifier layer 2 -> nc_pred ----
  gemm_bf16_wmma<0, true, false><<<dim3(cdiv(C, BN), cdiv(NL, BM)), 128, 0, stream>>>(
      h2_b, wgc4_t, nullptr, xw4, nullptr, NL, C, H, 0, 0);
  hipMemcpyAsync(ncpre, xw4, (size_t)NL * C * 4, hipMemcpyDeviceToDevice, stream);
  spmm_orig<<<cdiv(E, 8), 256, 0, stream>>>(edges, xw4, ncpre, E, C);
  spmm_new<<<cdiv(NP, 8), 256, 0, stream>>>(newval, xw4, ncpre, NP, P, Nn, C);
  cls_finalize<false><<<cdiv((size_t)NL * C, 256), 256, 0, stream>>>(
      ncpre, rowsum, b_gc4, nullptr, out_nc, (size_t)NL * C, C);
}
